// TransformerBlock3D_29265907155629
// MI455X (gfx1250) — matrix operand order repaired and checked by reading, not compiled, not run on hardware
//
#include <hip/hip_runtime.h>
#include <hip/hip_bf16.h>
#include <math.h>

// ---------------------------------------------------------------------------
// 3D Swin transformer block for MI455X (gfx1250, wave32, WMMA)
// GEMMs via v_wmma_f32_16x16x32_f16 with 2x4 register blocking per wave.
// ---------------------------------------------------------------------------

typedef _Float16 h8   __attribute__((ext_vector_type(8)));
typedef _Float16 v16h __attribute__((ext_vector_type(16)));
typedef float    v8f  __attribute__((ext_vector_type(8)));

#define DIM   192
#define NH    6
#define HEAD  32
#define HID   768
#define NWIN  98      // tokens per (2,7,7) window
#define NPAD  128     // padded token count for attention K-dim
#define BATCH 2
#define DD    16
#define HH    56
#define WW    56
#define NTOK  (DD*HH*WW)          // 50176 per batch
#define MROWS (BATCH*NTOK)        // 100352
#define BW    1024                // windows total (2*8*8*8)
#define SCALE_Q 0.17677669529663687f

// Build a 16-half fragment from two contiguous 8-half chunks.
__device__ __forceinline__ v16h make_frag(const _Float16* p0, const _Float16* p1) {
    union { v16h v; h8 h[2]; } u;
    u.h[0] = *(const h8*)p0;
    u.h[1] = *(const h8*)p1;
    return u.v;
}

__device__ __forceinline__ v16h load_frag_a(const _Float16* Arow, int k0, int halfSel) {
    return make_frag(Arow + k0 + halfSel * 8, Arow + k0 + halfSel * 8 + 16);
}
__device__ __forceinline__ v16h load_frag_b(const _Float16* Bcol, int k0, int halfSel) {
    return make_frag(Bcol + k0 + halfSel * 8, Bcol + k0 + halfSel * 8 + 16);
}
__device__ __forceinline__ v8f wmma16(v16h a, v16h b, v8f c) {
    return __builtin_amdgcn_wmma_f32_16x16x32_f16(false, a, false, b, (short)0, c, false, false);
}

// 32x64 per-wave GEMM core: 2 A-frags + 4 B-frags -> 8 WMMAs per k-step.
// m/n already include the per-lane (lane&15) offset.
__device__ __forceinline__ void gemm_2x4(const _Float16* __restrict__ A, int lda,
                                         const _Float16* __restrict__ BT, int ldb,
                                         int K, int m, int n, int halfSel,
                                         v8f acc[2][4]) {
    const _Float16* A0 = A + (size_t)m * lda;
    const _Float16* A1 = A0 + (size_t)16 * lda;
    const _Float16* B0 = BT + (size_t)n * ldb;
    for (int k0 = 0; k0 < K; k0 += 32) {
        v16h a0 = load_frag_a(A0, k0, halfSel);
        v16h a1 = load_frag_a(A1, k0, halfSel);
#pragma unroll
        for (int j = 0; j < 4; ++j) {
            v16h b = load_frag_b(B0 + (size_t)(j * 16) * ldb, k0, halfSel);
            acc[0][j] = wmma16(a0, b, acc[0][j]);
            acc[1][j] = wmma16(a1, b, acc[1][j]);
        }
    }
}

// ---------------------------------------------------------------------------
// Weight convert + transpose: w[K,N] f32 -> wT[N,K] f16
// ---------------------------------------------------------------------------
__global__ void k_transpose_w(const float* __restrict__ w, _Float16* __restrict__ wT,
                              int K, int N) {
    int idx = blockIdx.x * blockDim.x + threadIdx.x;
    if (idx >= K * N) return;
    int k = idx / N, n = idx % N;
    wT[(size_t)n * K + k] = (_Float16)w[idx];
}

// ---------------------------------------------------------------------------
// LayerNorm (192 ch). gather==1: write in window-partitioned order.
// ---------------------------------------------------------------------------
__global__ void k_layernorm(const float* __restrict__ x, const float* __restrict__ g,
                            const float* __restrict__ b, _Float16* __restrict__ out,
                            int gather) {
    int token = blockIdx.x;
    int c = threadIdx.x;
    const float* xr = x + (size_t)token * DIM;
    __shared__ float red[256];
    float v = (c < DIM) ? xr[c] : 0.f;
    red[c] = v; __syncthreads();
    for (int s = 128; s > 0; s >>= 1) { if (c < s) red[c] += red[c + s]; __syncthreads(); }
    float mu = red[0] * (1.f / DIM);
    __syncthreads();
    float dv = (c < DIM) ? (v - mu) : 0.f;
    red[c] = dv * dv; __syncthreads();
    for (int s = 128; s > 0; s >>= 1) { if (c < s) red[c] += red[c + s]; __syncthreads(); }
    float var = red[0] * (1.f / DIM);
    if (c >= DIM) return;
    float y = (v - mu) * rsqrtf(var + 1e-5f) * g[c] + b[c];
    size_t dst;
    if (gather) {
        int bb = token / NTOK, n = token % NTOK;
        int d = n / (HH * WW), rem = n % (HH * WW), h = rem / WW, w = rem % WW;
        int win = ((bb * 8 + d / 2) * 8 + h / 7) * 8 + w / 7;
        int t = (d % 2) * 49 + (h % 7) * 7 + (w % 7);
        dst = ((size_t)win * NWIN + t) * DIM + c;
    } else {
        dst = (size_t)token * DIM + c;
    }
    out[dst] = (_Float16)y;
}

// ---------------------------------------------------------------------------
// QKV GEMM: hwin[100352,192] @ qkvT[576,192] -> scatter into padded q/k/vT.
// ---------------------------------------------------------------------------
__global__ void k_gemm_qkv(const _Float16* __restrict__ A, const _Float16* __restrict__ BT,
                           const float* __restrict__ qkv_b,
                           _Float16* __restrict__ qbuf, _Float16* __restrict__ kbuf,
                           _Float16* __restrict__ vTb) {
    const int K = DIM, N = 3 * DIM;
    int wave = threadIdx.x >> 5, lane = threadIdx.x & 31;
    int halfSel = lane >> 4, r = lane & 15;
    long mt = (long)blockIdx.x * 8 + wave;
    int ntn = N / 64;
    int tm = (int)(mt / ntn), tn = (int)(mt % ntn);
    v8f acc[2][4];
    for (int jt = 0; jt < 2; ++jt) for (int j = 0; j < 4; ++j) acc[jt][j] = (v8f){};
    gemm_2x4(A, K, BT, K, K, tm * 32 + r, tn * 64 + r, halfSel, acc);

    int cj[4], whichj[4], headj[4], hcj[4];
    float biasj[4];
    for (int j = 0; j < 4; ++j) {
        int c = tn * 64 + j * 16 + r;
        cj[j] = c; biasj[j] = qkv_b[c];
        whichj[j] = c / DIM; headj[j] = (c / HEAD) % NH; hcj[j] = c % HEAD;
    }
    for (int jt = 0; jt < 2; ++jt)
        for (int i = 0; i < 8; ++i) {
            int row = tm * 32 + jt * 16 + i + halfSel * 8;
            int win = row / NWIN, t = row % NWIN;
            for (int j = 0; j < 4; ++j) {
                float val = acc[jt][j][i] + biasj[j];
                size_t base = (size_t)(win * NH + headj[j]);
                if (whichj[j] == 0)
                    qbuf[(base * NPAD + t) * HEAD + hcj[j]] = (_Float16)(val * SCALE_Q);
                else if (whichj[j] == 1)
                    kbuf[(base * NPAD + t) * HEAD + hcj[j]] = (_Float16)val;
                else
                    vTb[(base * HEAD + hcj[j]) * NPAD + t] = (_Float16)val;
            }
        }
}

// ---------------------------------------------------------------------------
// Attention: one block per (window,head). 7 waves; wave w owns rows [16w,16w+16).
// Q-fragment loaded once and reused over 7 S-tiles; softmax via 16-lane xor
// shuffles; P staged f16 in LDS; O = P @ vT with A-frag reuse over 2 N-tiles.
// ---------------------------------------------------------------------------
__global__ void k_attention(const _Float16* __restrict__ qbuf, const _Float16* __restrict__ kbuf,
                            const _Float16* __restrict__ vT, const float* __restrict__ bias_table,
                            _Float16* __restrict__ obuf) {
    int wh = blockIdx.x;
    int head = wh % NH, win = wh / NH;
    int wave = threadIdx.x >> 5, lane = threadIdx.x & 31;
    int halfSel = lane >> 4, r = lane & 15;
    const _Float16* q = qbuf + (size_t)wh * NPAD * HEAD;
    const _Float16* k = kbuf + (size_t)wh * NPAD * HEAD;
    const _Float16* v = vT  + (size_t)wh * HEAD * NPAD;
    __shared__ _Float16 P[112][NPAD];

    int m = wave * 16 + r;
    // S = q @ k^T : single k-step (HEAD=32); one A-frag reused over 7 tiles
    v16h aq = load_frag_a(q + m * HEAD, 0, halfSel);
    v8f acc[7];
#pragma unroll
    for (int t = 0; t < 7; ++t) {
        v16h b = load_frag_b(k + (t * 16 + r) * HEAD, 0, halfSel);
        acc[t] = wmma16(aq, b, (v8f){});
    }
    // bias + mask + softmax per held row
    for (int i = 0; i < 8; ++i) {
        int rowm = wave * 16 + i + halfSel * 8;
        int tt = (rowm < NWIN) ? rowm : 0;
        int d1 = tt / 49, rm1 = tt % 49, h1 = rm1 / 7, w1 = rm1 % 7;
        float vals[7];
        float vmax = -3.0e38f;
        for (int t = 0; t < 7; ++t) {
            int col = t * 16 + r;
            float s = -3.0e38f;
            if (col < NWIN) {
                int d2 = col / 49, rm2 = col % 49, h2 = rm2 / 7, w2 = rm2 % 7;
                int idx = (d1 - d2 + 1) * 169 + (h1 - h2 + 6) * 13 + (w1 - w2 + 6);
                s = acc[t][i] + bias_table[idx * NH + head];
            }
            vals[t] = s;
            vmax = fmaxf(vmax, s);
        }
        for (int msk = 1; msk < 16; msk <<= 1)
            vmax = fmaxf(vmax, __shfl_xor(vmax, msk, 32));
        float e[7], ssum = 0.f;
        for (int t = 0; t < 7; ++t) {
            e[t] = (t * 16 + r < NWIN) ? __expf(vals[t] - vmax) : 0.f;
            ssum += e[t];
        }
        for (int msk = 1; msk < 16; msk <<= 1)
            ssum += __shfl_xor(ssum, msk, 32);
        float inv = 1.f / ssum;
        for (int t = 0; t < 7; ++t) P[rowm][t * 16 + r] = (_Float16)(e[t] * inv);
        P[rowm][112 + r] = (_Float16)0.f;   // zero K-padding cols 112..127
    }
    __syncthreads();
    // O = P[112,128] @ vT : per k-step one LDS A-frag reused over 2 N-tiles
    v8f o0 = (v8f){}, o1 = (v8f){};
    const _Float16* b0 = v + (size_t)(0 + r) * NPAD;
    const _Float16* b1 = v + (size_t)(16 + r) * NPAD;
#pragma unroll
    for (int k0 = 0; k0 < NPAD; k0 += 32) {
        v16h a = load_frag_a(&P[m][0], k0, halfSel);
        o0 = wmma16(a, load_frag_b(b0, k0, halfSel), o0);
        o1 = wmma16(a, load_frag_b(b1, k0, halfSel), o1);
    }
    for (int i = 0; i < 8; ++i) {
        int rowm = wave * 16 + i + halfSel * 8;
        if (rowm < NWIN) {
            size_t dst = ((size_t)win * NWIN + rowm) * DIM + head * HEAD;
            obuf[dst + r]      = (_Float16)o0[i];
            obuf[dst + 16 + r] = (_Float16)o1[i];
        }
    }
}

// ---------------------------------------------------------------------------
// Proj GEMM + window-reverse + residual: x1 = x + 0.5*(obuf @ projT + b)
// ---------------------------------------------------------------------------
__global__ void k_gemm_proj(const _Float16* __restrict__ A, const _Float16* __restrict__ BT,
                            const float* __restrict__ proj_b, const float* __restrict__ x,
                            float* __restrict__ x1) {
    const int K = DIM, N = DIM;
    int wave = threadIdx.x >> 5, lane = threadIdx.x & 31;
    int halfSel = lane >> 4, r = lane & 15;
    long mt = (long)blockIdx.x * 8 + wave;
    int ntn = N / 64;
    int tm = (int)(mt / ntn), tn = (int)(mt % ntn);
    v8f acc[2][4];
    for (int jt = 0; jt < 2; ++jt) for (int j = 0; j < 4; ++j) acc[jt][j] = (v8f){};
    gemm_2x4(A, K, BT, K, K, tm * 32 + r, tn * 64 + r, halfSel, acc);

    float biasj[4];
    for (int j = 0; j < 4; ++j) biasj[j] = proj_b[tn * 64 + j * 16 + r];
    for (int jt = 0; jt < 2; ++jt)
        for (int i = 0; i < 8; ++i) {
            int row = tm * 32 + jt * 16 + i + halfSel * 8;
            int win = row / NWIN, t = row % NWIN;
            int bb = win >> 9, wd = (win >> 6) & 7, wh = (win >> 3) & 7, wwn = win & 7;
            int ind = t / 49, rem = t % 49, inh = rem / 7, inw = rem % 7;
            int d = wd * 2 + ind, h = wh * 7 + inh, w = wwn * 7 + inw;
            size_t bn = (size_t)bb * NTOK + d * (HH * WW) + h * WW + w;
            for (int j = 0; j < 4; ++j) {
                int c = tn * 64 + j * 16 + r;
                x1[bn * DIM + c] = x[bn * DIM + c] + 0.5f * (acc[jt][j][i] + biasj[j]);
            }
        }
}

// ---------------------------------------------------------------------------
// fc1 GEMM: h2[100352,192] @ fc1T[768,192] -> ffn1 f16 NCDHW [2,768,50176]
// ---------------------------------------------------------------------------
__global__ void k_gemm_fc1(const _Float16* __restrict__ A, const _Float16* __restrict__ BT,
                           const float* __restrict__ fc1_b, _Float16* __restrict__ ffn1) {
    const int K = DIM, N = HID;
    int wave = threadIdx.x >> 5, lane = threadIdx.x & 31;
    int halfSel = lane >> 4, r = lane & 15;
    long mt = (long)blockIdx.x * 8 + wave;
    int ntn = N / 64;
    int tm = (int)(mt / ntn), tn = (int)(mt % ntn);
    v8f acc[2][4];
    for (int jt = 0; jt < 2; ++jt) for (int j = 0; j < 4; ++j) acc[jt][j] = (v8f){};
    gemm_2x4(A, K, BT, K, K, tm * 32 + r, tn * 64 + r, halfSel, acc);

    float biasj[4];
    for (int j = 0; j < 4; ++j) biasj[j] = fc1_b[tn * 64 + j * 16 + r];
    for (int jt = 0; jt < 2; ++jt)
        for (int i = 0; i < 8; ++i) {
            int row = tm * 32 + jt * 16 + i + halfSel * 8;
            int bb = row / NTOK, nn = row % NTOK;
            for (int j = 0; j < 4; ++j) {
                int c = tn * 64 + j * 16 + r;
                ffn1[((size_t)(bb * HID + c)) * NTOK + nn] = (_Float16)(acc[jt][j][i] + biasj[j]);
            }
        }
}

// ---------------------------------------------------------------------------
// Depthwise 3x3x3 conv + exact GELU: ffn1 NCDHW -> h3 [bn,768] f16
// ---------------------------------------------------------------------------
__global__ void k_dwconv_gelu(const _Float16* __restrict__ ffn1, const float* __restrict__ dw_w,
                              const float* __restrict__ dw_b, _Float16* __restrict__ h3) {
    long idx = (long)blockIdx.x * blockDim.x + threadIdx.x;
    if (idx >= (long)BATCH * HID * NTOK) return;
    int n = (int)(idx % NTOK);
    int c = (int)((idx / NTOK) % HID);
    int bb = (int)(idx / ((long)NTOK * HID));
    int d = n / (HH * WW), rem = n % (HH * WW), h = rem / WW, w = rem % WW;
    const _Float16* src = ffn1 + (size_t)(bb * HID + c) * NTOK;
    const float* wc = dw_w + (size_t)c * 27;
    float s = 0.f;
    for (int kd = -1; kd <= 1; ++kd) {
        int dz = d + kd; if (dz < 0 || dz >= DD) continue;
        for (int kh = -1; kh <= 1; ++kh) {
            int hy = h + kh; if (hy < 0 || hy >= HH) continue;
            for (int kw = -1; kw <= 1; ++kw) {
                int wx = w + kw; if (wx < 0 || wx >= WW) continue;
                float val = (float)src[dz * (HH * WW) + hy * WW + wx];
                s += val * wc[(kd + 1) * 9 + (kh + 1) * 3 + (kw + 1)];
            }
        }
    }
    s += dw_b[c];
    float g = 0.5f * s * (1.f + erff(s * 0.70710678118654752f));
    h3[((size_t)bb * NTOK + n) * HID + c] = (_Float16)g;
}

// ---------------------------------------------------------------------------
// fc2 GEMM + residual: out = x1 + 0.5*(h3 @ fc2T + b)
// ---------------------------------------------------------------------------
__global__ void k_gemm_fc2(const _Float16* __restrict__ A, const _Float16* __restrict__ BT,
                           const float* __restrict__ fc2_b, const float* __restrict__ x1,
                           float* __restrict__ out) {
    const int K = HID, N = DIM;
    int wave = threadIdx.x >> 5, lane = threadIdx.x & 31;
    int halfSel = lane >> 4, r = lane & 15;
    long mt = (long)blockIdx.x * 8 + wave;
    int ntn = N / 64;
    int tm = (int)(mt / ntn), tn = (int)(mt % ntn);
    v8f acc[2][4];
    for (int jt = 0; jt < 2; ++jt) for (int j = 0; j < 4; ++j) acc[jt][j] = (v8f){};
    gemm_2x4(A, K, BT, K, K, tm * 32 + r, tn * 64 + r, halfSel, acc);

    float biasj[4];
    for (int j = 0; j < 4; ++j) biasj[j] = fc2_b[tn * 64 + j * 16 + r];
    for (int jt = 0; jt < 2; ++jt)
        for (int i = 0; i < 8; ++i) {
            size_t row = (size_t)(tm * 32 + jt * 16 + i + halfSel * 8);
            for (int j = 0; j < 4; ++j) {
                int c = tn * 64 + j * 16 + r;
                out[row * DIM + c] = x1[row * DIM + c] + 0.5f * (acc[jt][j][i] + biasj[j]);
            }
        }
}

// ---------------------------------------------------------------------------
extern "C" void kernel_launch(void* const* d_in, const int* in_sizes, int n_in,
                              void* d_out, int out_size, void* d_ws, size_t ws_size,
                              hipStream_t stream) {
    const float* x          = (const float*)d_in[0];
    const float* norm1_g    = (const float*)d_in[1];
    const float* norm1_b    = (const float*)d_in[2];
    const float* qkv_w      = (const float*)d_in[3];
    const float* qkv_b      = (const float*)d_in[4];
    const float* bias_table = (const float*)d_in[5];
    const float* proj_w     = (const float*)d_in[6];
    const float* proj_b     = (const float*)d_in[7];
    const float* norm2_g    = (const float*)d_in[8];
    const float* norm2_b    = (const float*)d_in[9];
    const float* fc1_w      = (const float*)d_in[10];
    const float* fc1_b      = (const float*)d_in[11];
    const float* dw_w       = (const float*)d_in[12];
    const float* dw_b       = (const float*)d_in[13];
    const float* fc2_w      = (const float*)d_in[14];
    const float* fc2_b      = (const float*)d_in[15];
    float* out = (float*)d_out;

    // --- scratch bump allocator (lifetime-aliased regions) ---
    char* ws = (char*)d_ws;
    size_t off = 0;
    auto alloc = [&](size_t bytes) -> char* {
        char* p = ws + off;
        off = (off + bytes + 255) & ~(size_t)255;
        return p;
    };
    _Float16* qkvT  = (_Float16*)alloc((size_t)576 * 192 * 2);
    _Float16* projT = (_Float16*)alloc((size_t)192 * 192 * 2);
    _Float16* fc1T  = (_Float16*)alloc((size_t)768 * 192 * 2);
    _Float16* fc2T  = (_Float16*)alloc((size_t)192 * 768 * 2);
    float*    x1    = (float*)   alloc((size_t)MROWS * DIM * 4);
    // R1: hwin (LN1 out, dead after qkv GEMM) aliased with ffn1 (fc1 out)
    char* R1 = alloc((size_t)BATCH * HID * NTOK * 2);
    _Float16* hwin = (_Float16*)R1;
    _Float16* ffn1 = (_Float16*)R1;
    // R2: q/k/vT (dead after attention) aliased with h3 (conv out)
    size_t qkvsz = (size_t)BW * NH * NPAD * HEAD * 2;
    char* R2 = alloc(3 * qkvsz > (size_t)MROWS * HID * 2 ? 3 * qkvsz : (size_t)MROWS * HID * 2);
    _Float16* qbuf = (_Float16*)R2;
    _Float16* kbuf = (_Float16*)(R2 + qkvsz);
    _Float16* vTb  = (_Float16*)(R2 + 2 * qkvsz);
    _Float16* h3   = (_Float16*)R2;
    // R3: obuf (dead after proj GEMM) aliased with h2 (LN2 out)
    char* R3 = alloc((size_t)MROWS * DIM * 2);
    _Float16* obuf = (_Float16*)R3;
    _Float16* h2   = (_Float16*)R3;

    // --- 0) weight transpose + f16 convert ---
    k_transpose_w<<<(192 * 576 + 255) / 256, 256, 0, stream>>>(qkv_w, qkvT, 192, 576);
    k_transpose_w<<<(192 * 192 + 255) / 256, 256, 0, stream>>>(proj_w, projT, 192, 192);
    k_transpose_w<<<(192 * 768 + 255) / 256, 256, 0, stream>>>(fc1_w, fc1T, 192, 768);
    k_transpose_w<<<(768 * 192 + 255) / 256, 256, 0, stream>>>(fc2_w, fc2T, 768, 192);

    // --- 1) LN1 + window partition ---
    k_layernorm<<<MROWS, 256, 0, stream>>>(x, norm1_g, norm1_b, hwin, 1);

    // --- 2) QKV GEMM (32x64 macro-tiles, 8 waves/block) ---
    k_gemm_qkv<<<(MROWS / 32) * (576 / 64) / 8, 256, 0, stream>>>(hwin, qkvT, qkv_b,
                                                                  qbuf, kbuf, vTb);
    // --- 3) attention per (window, head) ---
    k_attention<<<BW * NH, 224, 0, stream>>>(qbuf, kbuf, vTb, bias_table, obuf);

    // --- 4) proj GEMM + window reverse + residual ---
    k_gemm_proj<<<(MROWS / 32) * (192 / 64) / 8, 256, 0, stream>>>(obuf, projT, proj_b, x, x1);

    // --- 5) LN2 ---
    k_layernorm<<<MROWS, 256, 0, stream>>>(x1, norm2_g, norm2_b, h2, 0);

    // --- 6) fc1 GEMM -> NCDHW f16 ---
    k_gemm_fc1<<<(MROWS / 32) * (HID / 64) / 8, 256, 0, stream>>>(h2, fc1T, fc1_b, ffn1);

    // --- 7) depthwise conv + GELU ---
    long nconv = (long)BATCH * HID * NTOK;
    k_dwconv_gelu<<<(int)((nconv + 255) / 256), 256, 0, stream>>>(ffn1, dw_w, dw_b, h3);

    // --- 8) fc2 GEMM + residual -> out ---
    k_gemm_fc2<<<(MROWS / 32) * (192 / 64) / 8, 256, 0, stream>>>(h3, fc2T, fc2_b, x1, out);

    (void)in_sizes; (void)n_in; (void)out_size; (void)ws_size;
}
